// plga_layer_81123342287392
// MI455X (gfx1250) — matrix operand order repaired and checked by reading, not compiled, not run on hardware
//
#include <hip/hip_runtime.h>

// CDNA5 / gfx1250 flash-attention with per-head query projection.
// v_wmma_f32_16x16x32_bf16 (wave32); 32 query rows per wave; base-2
// register-resident online softmax (row max via lane shuffles, row sums via
// P @ ones WMMA, raw v_exp_f32 via __builtin_amdgcn_exp2f); double-buffered
// K/V staging with register prefetch + global_prefetch of tile j+2.
// K fragments are scoped per row-group to keep VGPR usage under 256.

typedef __attribute__((ext_vector_type(16))) __bf16 v16bf;
typedef __attribute__((ext_vector_type(8)))  __bf16 v8bf;
typedef __attribute__((ext_vector_type(4)))  __bf16 v4bf;
typedef __attribute__((ext_vector_type(8)))  float  v8f;
typedef __attribute__((ext_vector_type(4)))  float  v4f;

namespace {

constexpr int S_LEN = 2048;
constexpr int D_DIM = 64;
constexpr int H_NUM = 16;
constexpr int BM    = 128;  // query rows per workgroup (4 waves x 32)
constexpr int BN    = 32;   // key columns per iteration
// LDS row strides (elements), padded against bank conflicts, 16B aligned
constexpr int QSTR = 72;    // bf16
constexpr int ASTR = 72;    // bf16
constexpr int KSTR = 72;    // bf16
constexpr int VSTR = 40;    // bf16 (transposed V: [d][key])
constexpr int PSTR = 40;    // bf16 probabilities

// Q scale: 1/sqrt(D) * log2(e)  -> softmax runs in base-2 domain (v_exp_f32
// is natively base-2).
constexpr float QSCALE = 0.125f * 1.44269504088896340736f;

union AFrag { v16bf v; v8bf h[2]; };
union CAcc  { v8f   v; float f[8]; };

__device__ inline v4bf cvt4(v4f x) {
  v4bf y = { (__bf16)x.x, (__bf16)x.y, (__bf16)x.z, (__bf16)x.w };
  return y;
}

struct KVRegs { v4f k[4]; v4f v[4]; };

__device__ inline KVRegs load_kv(const float* __restrict__ hkB,
                                 const float* __restrict__ hvB,
                                 int kv0, int tid) {
  KVRegs r;
#pragma unroll
  for (int i = 0; i < 4; ++i) {
    int f4 = tid + i * 128;            // 512 float4's = 32x64 tile
    int kr = f4 >> 4;
    int c4 = (f4 & 15) << 2;
    r.k[i] = *(const v4f*)(hkB + (size_t)(kv0 + kr) * D_DIM + c4);
    r.v[i] = *(const v4f*)(hvB + (size_t)(kv0 + kr) * D_DIM + c4);
  }
  return r;
}

__device__ inline void store_kv(const KVRegs& r, __bf16 (*sK)[KSTR],
                                __bf16 (*sVt)[VSTR], int tid) {
#pragma unroll
  for (int i = 0; i < 4; ++i) {
    int f4 = tid + i * 128;
    int kr = f4 >> 4;
    int c4 = (f4 & 15) << 2;
    *(v4bf*)&sK[kr][c4] = cvt4(r.k[i]);
    sVt[c4 + 0][kr] = (__bf16)r.v[i].x;   // V staged transposed [d][key]
    sVt[c4 + 1][kr] = (__bf16)r.v[i].y;
    sVt[c4 + 2][kr] = (__bf16)r.v[i].z;
    sVt[c4 + 3][kr] = (__bf16)r.v[i].w;
  }
}

} // namespace

__global__ __launch_bounds__(128) void fa_wmma_kernel(
    const float* __restrict__ Hin, const float* __restrict__ Hk,
    const float* __restrict__ Hv,  const float* __restrict__ avAp,
    float* __restrict__ out)
{
  __shared__ __attribute__((aligned(16))) __bf16 sQ [BM][QSTR];      // Hin, then Q
  __shared__ __attribute__((aligned(16))) __bf16 sAT[D_DIM][ASTR];   // avAp^T [e][d]
  __shared__ __attribute__((aligned(16))) __bf16 sK [2][BN][KSTR];   // K row-major
  __shared__ __attribute__((aligned(16))) __bf16 sVt[2][D_DIM][VSTR];// V^T [d][key]
  __shared__ __attribute__((aligned(16))) __bf16 sP [4][32][PSTR];   // probs/wave

  const int tid  = threadIdx.x;
  const int wave = tid >> 5;
  const int lane = tid & 31;
  const int ln   = lane & 15;
  const int hi   = lane >> 4;

  const int qb = blockIdx.x & 15;          // S/BM = 16 query blocks
  const int bh = blockIdx.x >> 4;          // b*H + h
  const int h  = bh & (H_NUM - 1);
  const int q0 = qb * BM;
  const int r0 = wave * 32;                // wave's row base inside block

  const float* hinB = Hin  + (size_t)bh * S_LEN * D_DIM;
  const float* hkB  = Hk   + (size_t)bh * S_LEN * D_DIM;
  const float* hvB  = Hv   + (size_t)bh * S_LEN * D_DIM;
  const float* apB  = avAp + (size_t)h  * D_DIM * D_DIM;
  float*       outB = out  + (size_t)bh * S_LEN * D_DIM;

  // Issue K/V tile-0 global loads early: latency hides under Q projection.
  KVRegs kv = load_kv(hkB, hvB, 0, tid);

  // ---- stage Hin tile (bf16 row-major): 128x64 ----
#pragma unroll
  for (int i = 0; i < 16; ++i) {
    int f4  = tid + i * 128;               // 2048 float4's
    int row = f4 >> 4;
    int c4  = (f4 & 15) << 2;
    v4f x = *(const v4f*)(hinB + (size_t)(q0 + row) * D_DIM + c4);
    *(v4bf*)&sQ[row][c4] = cvt4(x);
  }
  // ---- stage avAp^T (bf16): 64x64 ----
#pragma unroll
  for (int i = 0; i < 8; ++i) {
    int f4  = tid + i * 128;               // 1024 float4's
    int row = f4 >> 4;
    int c4  = (f4 & 15) << 2;
    v4f a = *(const v4f*)(apB + (size_t)row * D_DIM + c4);
    sAT[c4 + 0][row] = (__bf16)a.x;
    sAT[c4 + 1][row] = (__bf16)a.y;
    sAT[c4 + 2][row] = (__bf16)a.z;
    sAT[c4 + 3][row] = (__bf16)a.w;
  }
  __syncthreads();

  // ---- Q = Hin @ avAp (per wave 32x64; avAp fragments reused by both
  //      row-groups) ----
  {
    AFrag ah[2][2];
#pragma unroll
    for (int g = 0; g < 2; ++g)
#pragma unroll
      for (int c = 0; c < 2; ++c) {
        ah[g][c].h[0] = *(const v8bf*)&sQ[r0 + g * 16 + ln][c * 32 + hi * 8];
        ah[g][c].h[1] = *(const v8bf*)&sQ[r0 + g * 16 + ln][c * 32 + 16 + hi * 8];
      }
#pragma unroll
    for (int n = 0; n < 4; ++n) {
      CAcc qa[2];
#pragma unroll
      for (int g = 0; g < 2; ++g)
#pragma unroll
        for (int k = 0; k < 8; ++k) qa[g].f[k] = 0.0f;
#pragma unroll
      for (int c = 0; c < 2; ++c) {
        AFrag bm;
        bm.h[0] = *(const v8bf*)&sAT[n * 16 + ln][c * 32 + hi * 8];
        bm.h[1] = *(const v8bf*)&sAT[n * 16 + ln][c * 32 + 16 + hi * 8];
#pragma unroll
        for (int g = 0; g < 2; ++g)
          qa[g].v = __builtin_amdgcn_wmma_f32_16x16x32_bf16(
              false, ah[g][c].v, false, bm.v, (short)0, qa[g].v, false, false);
      }
      // fold QSCALE; C layout -> row-major LDS (own rows only)
#pragma unroll
      for (int g = 0; g < 2; ++g)
#pragma unroll
        for (int r = 0; r < 8; ++r)
          sQ[r0 + g * 16 + r + hi * 8][n * 16 + ln] =
              (__bf16)(qa[g].f[r] * QSCALE);
    }
  }

  // Reload Q as A fragments (wave-local rows; per-wave DS ops are in-order)
  AFrag aq[2][2];
#pragma unroll
  for (int g = 0; g < 2; ++g)
#pragma unroll
    for (int c = 0; c < 2; ++c) {
      aq[g][c].h[0] = *(const v8bf*)&sQ[r0 + g * 16 + ln][c * 32 + hi * 8];
      aq[g][c].h[1] = *(const v8bf*)&sQ[r0 + g * 16 + ln][c * 32 + 16 + hi * 8];
    }

  // Commit K/V tile 0 into buffer 0.
  store_kv(kv, sK[0], sVt[0], tid);
  __syncthreads();

  // All-ones B fragment for WMMA row-sum of P.
  AFrag ones;
#pragma unroll
  for (int i = 0; i < 16; ++i) ones.v[i] = (__bf16)1.0f;

  CAcc o[2][4];
#pragma unroll
  for (int g = 0; g < 2; ++g)
#pragma unroll
    for (int n = 0; n < 4; ++n)
#pragma unroll
      for (int k = 0; k < 8; ++k) o[g][n].f[k] = 0.0f;

  float m_run[2][8], l_run[2][8];          // replicated per 16-lane half
#pragma unroll
  for (int g = 0; g < 2; ++g)
#pragma unroll
    for (int r = 0; r < 8; ++r) { m_run[g][r] = -1.0e30f; l_run[g][r] = 0.0f; }

  const int wq_max = q0 + r0 + 31;         // wave's largest query row
  const int nj = (q0 >> 5) + 4;            // causal tile bound (keys <= q0+127)

#pragma unroll 1
  for (int j = 0; j < nj; ++j) {
    const int kv0 = j * BN;
    const int buf = j & 1;
    const bool hasNext = (j + 1) < nj;

    // Prefetch next tile into registers (overlaps with this tile's compute).
    if (hasNext) kv = load_kv(hkB, hvB, kv0 + BN, tid);
    // Warm L2/L0 two tiles ahead (global_prefetch_b8).
    if (j + 2 < nj) {
      const size_t off = (size_t)(kv0 + 2 * BN + (tid >> 2)) * D_DIM +
                         ((tid & 3) << 4);
      __builtin_prefetch(hkB + off, 0, 1);
      __builtin_prefetch(hvB + off, 0, 1);
    }

    if (kv0 <= wq_max) {                   // wave-uniform causal skip
      // Shared V fragments: loaded once, reused by both row-groups' PV.
      AFrag bv[4];
#pragma unroll
      for (int n = 0; n < 4; ++n) {
        bv[n].h[0] = *(const v8bf*)&sVt[buf][n * 16 + ln][hi * 8];
        bv[n].h[1] = *(const v8bf*)&sVt[buf][n * 16 + ln][16 + hi * 8];
      }

#pragma unroll
      for (int g = 0; g < 2; ++g) {
        if (kv0 > q0 + r0 + g * 16 + 15) continue;  // row-group fully masked

        // ---- S-tile = Q @ K^T (16 x 32), scores in log2 domain ----
        // K fragments scoped per row-group: short live range keeps the
        // kernel under 256 VGPRs (no s_set_vgpr_msb traffic).
        CAcc sc[2];
#pragma unroll
        for (int ct = 0; ct < 2; ++ct) {
#pragma unroll
          for (int k = 0; k < 8; ++k) sc[ct].f[k] = 0.0f;
#pragma unroll
          for (int c = 0; c < 2; ++c) {
            AFrag bk;
            bk.h[0] = *(const v8bf*)&sK[buf][ct * 16 + ln][c * 32 + hi * 8];
            bk.h[1] =
                *(const v8bf*)&sK[buf][ct * 16 + ln][c * 32 + 16 + hi * 8];
            sc[ct].v = __builtin_amdgcn_wmma_f32_16x16x32_bf16(
                false, aq[g][c].v, false, bk.v, (short)0, sc[ct].v,
                false, false);
          }
        }

        // ---- base-2 online softmax ----
        // C layout: lane holds column (kv0 + ct*16 + ln) of rows r + 8*hi.
        // xor masks 1/2/4/8 reduce within each 16-lane half (rows unmixed).
        float rs[8];
#pragma unroll
        for (int r = 0; r < 8; ++r) {
          const int grow = q0 + r0 + g * 16 + r + hi * 8;
          const bool ok0 = (kv0 + ln)      <= grow;
          const bool ok1 = (kv0 + 16 + ln) <= grow;
          float x0 = ok0 ? sc[0].f[r] : -1.0e30f;
          float x1 = ok1 ? sc[1].f[r] : -1.0e30f;
          float t = fmaxf(x0, x1);
          t = fmaxf(t, __shfl_xor(t, 1));
          t = fmaxf(t, __shfl_xor(t, 2));
          t = fmaxf(t, __shfl_xor(t, 4));
          t = fmaxf(t, __shfl_xor(t, 8));
          const float mn = fmaxf(m_run[g][r], t);
          // raw v_exp_f32; huge-negative args flush to 0 (wanted).
          rs[r] = __builtin_amdgcn_exp2f(m_run[g][r] - mn);
          float p0 = ok0 ? __builtin_amdgcn_exp2f(x0 - mn) : 0.0f;
          float p1 = ok1 ? __builtin_amdgcn_exp2f(x1 - mn) : 0.0f;
          sP[wave][g * 16 + r + hi * 8][ln]      = (__bf16)p0;
          sP[wave][g * 16 + r + hi * 8][16 + ln] = (__bf16)p1;
          m_run[g][r] = mn;
        }

        // ---- rescale O accumulators (factors already in registers) ----
#pragma unroll
        for (int n = 0; n < 4; ++n)
#pragma unroll
          for (int r = 0; r < 8; ++r) o[g][n].f[r] *= rs[r];

        // ---- P fragment (per-wave DS in-order covers sP) ----
        AFrag ap;
        ap.h[0] = *(const v8bf*)&sP[wave][g * 16 + ln][hi * 8];
        ap.h[1] = *(const v8bf*)&sP[wave][g * 16 + ln][16 + hi * 8];

        // ---- row sums of P via WMMA against all-ones B: every lane gets
        //      rowsum(row r+8*hi) in f[r] (replicated layout of l_run) ----
        CAcc sums;
#pragma unroll
        for (int k = 0; k < 8; ++k) sums.f[k] = 0.0f;
        sums.v = __builtin_amdgcn_wmma_f32_16x16x32_bf16(
            false, ap.v, false, ones.v, (short)0, sums.v, false, false);
#pragma unroll
        for (int r = 0; r < 8; ++r)
          l_run[g][r] = l_run[g][r] * rs[r] + sums.f[r];

        // ---- O += P @ V ----
#pragma unroll
        for (int n = 0; n < 4; ++n)
          o[g][n].v = __builtin_amdgcn_wmma_f32_16x16x32_bf16(
              false, ap.v, false, bv[n].v, (short)0, o[g][n].v, false, false);
      }
    }

    // Commit next tile into the buffer nobody reads this iteration; the
    // single barrier orders these writes against next iteration's reads.
    if (hasNext) store_kv(kv, sK[buf ^ 1], sVt[buf ^ 1], tid);
    __syncthreads();
  }

  // ---- epilogue: O /= l, store fp32 ----
#pragma unroll
  for (int g = 0; g < 2; ++g) {
    float linv[8];
#pragma unroll
    for (int r = 0; r < 8; ++r)
      linv[r] = (l_run[g][r] > 0.0f) ? (1.0f / l_run[g][r]) : 0.0f;
#pragma unroll
    for (int n = 0; n < 4; ++n)
#pragma unroll
      for (int r = 0; r < 8; ++r)
        outB[(size_t)(q0 + r0 + g * 16 + r + hi * 8) * D_DIM + n * 16 + ln] =
            o[g][n].f[r] * linv[r];
  }
}

extern "C" void kernel_launch(void* const* d_in, const int* in_sizes, int n_in,
                              void* d_out, int out_size, void* d_ws, size_t ws_size,
                              hipStream_t stream) {
  (void)in_sizes; (void)n_in; (void)d_ws; (void)ws_size; (void)out_size;
  const float* Hin  = (const float*)d_in[0];
  const float* Hk   = (const float*)d_in[1];
  const float* Hv   = (const float*)d_in[2];
  // d_in[3] = mask (causality computed analytically), d_in[5] = F_hidden
  const float* avAp = (const float*)d_in[4];
  float* out = (float*)d_out;

  dim3 grid((S_LEN / BM) * 2 * H_NUM);   // 16 q-blocks * 32 (b,h) = 512
  dim3 block(128);                        // 4 waves (wave32)
  fa_wmma_kernel<<<grid, block, 0, stream>>>(Hin, Hk, Hv, avAp, out);
}
